// Text2MelSimplified_60060822667530
// MI455X (gfx1250) — matrix operand order repaired and checked by reading, not compiled, not run on hardware
//
#include <hip/hip_runtime.h>
#include <stdint.h>

// ---------------- model dims ----------------
constexpr int cS     = 512;    // encoder length
constexpr int cDENC  = 1152;   // 512+512+128
constexpr int cWATT  = 128;
constexpr int cWIN   = 14;
constexpr int cTDEC  = 512;
constexpr int cMGC   = 80;
constexpr int cPROJ  = 100;
constexpr int cDIN   = 1252;
constexpr int cTPN   = 1536;   // TDEC*PF
constexpr int cPNC   = 256;
constexpr int cKP1   = 416;    // conv1 K (400) padded to mult of 32

constexpr int NBDEC  = 8;      // decoder persistent blocks
constexpr int TBDEC  = 512;    // decoder block size

typedef __attribute__((ext_vector_type(16))) __bf16          v16bf;
typedef __attribute__((ext_vector_type(16))) unsigned short  v16us;
typedef __attribute__((ext_vector_type(8)))  float           v8f;

__device__ __forceinline__ unsigned short f2bf(float f) {
  union { float f; unsigned int u; } x; x.f = f;
  unsigned int u = x.u;
  return (unsigned short)((u + 0x7FFFu + ((u >> 16) & 1u)) >> 16); // RNE
}
__device__ __forceinline__ float sigm(float x) { return 1.0f / (1.0f + __expf(-x)); }

// b128 dot product: n4 float4's
__device__ __forceinline__ float dot4(const float* __restrict__ w,
                                      const float* __restrict__ v, int n4) {
  const float4* w4 = (const float4*)w;
  const float4* v4 = (const float4*)v;
  float acc = 0.f;
  for (int k = 0; k < n4; ++k) {
    float4 a = w4[k], b = v4[k];
    acc += a.x * b.x + a.y * b.y + a.z * b.z + a.w * b.w;
  }
  return acc;
}

// ---------------------------------------------------------------------------
// bf16 WMMA GEMM:  C(MxN) = act( A(MxK,row) @ Bt(NxK,row)^T + bias[M] )
// K must be a multiple of 32; A rows and Bt rows 16B-aligned (K mult of 8).
// One wave32 per 16x16 C tile. Hot loop: 4x global_load_b128 + v_wmma.
// ---------------------------------------------------------------------------
__global__ void wmma_gemm_bf16(const unsigned short* __restrict__ A,
                               const unsigned short* __restrict__ Bt,
                               const float* __restrict__ bias,
                               float* __restrict__ Cf,
                               unsigned short* __restrict__ Cb,
                               int M, int N, int K, int act)
{
  const int mt   = blockIdx.x;
  const int nt   = blockIdx.y;
  const int lane = threadIdx.x;      // wave32
  const int half = lane >> 4;
  const int l16  = lane & 15;
  const unsigned short* arow = A  + (size_t)(mt * 16 + l16) * K;
  const unsigned short* brow = Bt + (size_t)(nt * 16 + l16) * K;
  v8f acc = {};
#pragma unroll 2
  for (int k0 = 0; k0 < K; k0 += 32) {
    // A frag: lane covers K = k0+half*8+[0..7] and k0+16+half*8+[0..7]
    union { uint4 q[2]; v16us v; } ua, ub;
    ua.q[0] = *(const uint4*)(arow + k0 + half * 8);
    ua.q[1] = *(const uint4*)(arow + k0 + 16 + half * 8);
    // B frag: lane covers K = k0+half*8+[0..7] and k0+16+half*8+[0..7] at fixed N
    ub.q[0] = *(const uint4*)(brow + k0 + half * 8);
    ub.q[1] = *(const uint4*)(brow + k0 + 16 + half * 8);
    v16bf a = __builtin_bit_cast(v16bf, ua.v);
    v16bf b = __builtin_bit_cast(v16bf, ub.v);
    acc = __builtin_amdgcn_wmma_f32_16x16x32_bf16(false, a, false, b,
                                                  (short)0, acc, false, false);
  }
  // C tile: M = mt*16 + half*8 + r, N = nt*16 + l16
#pragma unroll
  for (int r = 0; r < 8; ++r) {
    int gm = mt * 16 + half * 8 + r;
    int gn = nt * 16 + l16;
    float v = acc[r];
    if (bias) v += bias[gm];
    if (act == 1) v = tanhf(v);
    if (Cf) Cf[(size_t)gm * N + gn] = v;
    if (Cb) Cb[(size_t)gm * N + gn] = f2bf(v);
  }
}

// ---------------- prep kernels ----------------
__global__ void k_cvt_bf16(const float* __restrict__ s, unsigned short* __restrict__ d, int n) {
  int i = blockIdx.x * blockDim.x + threadIdx.x;
  if (i < n) d[i] = f2bf(s[i]);
}
// convert with K -> KP zero padding per row
__global__ void k_cvt_pad(const float* __restrict__ s, unsigned short* __restrict__ d,
                          int M, int K, int KP) {
  int i = blockIdx.x * blockDim.x + threadIdx.x;
  if (i >= M * KP) return;
  int m = i / KP, k = i % KP;
  d[i] = (k < K) ? f2bf(s[(size_t)m * K + k]) : (unsigned short)0;
}
__global__ void k_zero_i32(int* __restrict__ p, int n) {
  int i = blockIdx.x * blockDim.x + threadIdx.x;
  if (i < n) p[i] = 0;
}

__global__ void k_styspk(const float* __restrict__ styp, const float* __restrict__ gst,
                         const float* __restrict__ semb, const int* __restrict__ spkid,
                         float* __restrict__ styspk) {
  int d = threadIdx.x; // 128
  float s = 0.f;
  for (int g = 0; g < 16; ++g) s += styp[g] * tanhf(gst[g * 128 + d]);
  styspk[d] = s + semb[spkid[0] * 128 + d];
}

// xs row-major (t, f): serves directly as Bt (N=512 rows, K=128)
__global__ void k_build_xs(const int* __restrict__ x, const int* __restrict__ xc,
                           const float* __restrict__ cemb, const float* __restrict__ caemb,
                           unsigned short* __restrict__ xs) {
  int i = blockIdx.x * blockDim.x + threadIdx.x; // 512*128
  if (i >= cS * 128) return;
  int t = i / 128, f = i % 128;
  float v = (f < 100) ? cemb[x[t] * 100 + f] : caemb[xc[t] * 28 + (f - 100)];
  xs[i] = f2bf(v);
}

__global__ void k_build_enc(const float* __restrict__ hf, const float* __restrict__ hb,
                            const float* __restrict__ styspk,
                            float* __restrict__ enc, unsigned short* __restrict__ enc_bf) {
  int i = blockIdx.x * blockDim.x + threadIdx.x;
  if (i >= cS * cDENC) return;
  int t = i / cDENC, j = i % cDENC;
  float v;
  if (j < 512)       v = hf[(size_t)t * 512 + j];
  else if (j < 1024) v = hb[(size_t)t * 512 + (j - 512)];
  else               v = styspk[j - 1024];
  enc[i] = v;
  enc_bf[i] = f2bf(v);
}

// im2col^T for k=5 SAME conv: Xcol (T x KP), Xcol[t][c*5+kk] = x[c][t+kk-2]
__global__ void k_im2col(const unsigned short* __restrict__ x /* C x T bf16 */,
                         unsigned short* __restrict__ Xcol, int C, int T, int KP) {
  int i = blockIdx.x * blockDim.x + threadIdx.x;
  if (i >= T * KP) return;
  int t = i / KP, k = i % KP;
  unsigned short v = 0;
  if (k < C * 5) {
    int c = k / 5, kk = k - c * 5;
    int tt = t + kk - 2;
    if (tt >= 0 && tt < T) v = x[(size_t)c * T + tt];
  }
  Xcol[i] = v;
}

// ---------------- encoder recurrence (Zih precomputed by WMMA) --------------
__global__ __launch_bounds__(1024) void k_enc_scan(const float* __restrict__ Whh_f,
                                                   const float* __restrict__ Whh_b,
                                                   const float* __restrict__ Zf,
                                                   const float* __restrict__ Zb,
                                                   float* __restrict__ hf,
                                                   float* __restrict__ hb) {
  __shared__ float h[512], c[512], zg[2048];
  const int dir = blockIdx.x;
  const float* Whh = dir ? Whh_b : Whh_f;
  const float* Z   = dir ? Zb : Zf;
  float* hs        = dir ? hb : hf;
  const int tid = threadIdx.x;
  if (tid < 512) { h[tid] = 0.f; c[tid] = 0.f; }
  __syncthreads();
  for (int s = 0; s < cS; ++s) {
    int t = dir ? (cS - 1 - s) : s;
#pragma unroll
    for (int rr = 0; rr < 2; ++rr) {
      int r = tid + rr * 1024;
      zg[r] = Z[(size_t)r * cS + t] + dot4(Whh + (size_t)r * 512, h, 128);
    }
    __syncthreads();
    if (tid < 512) {
      float zi = zg[tid], zff = zg[512 + tid], zgg = zg[1024 + tid], zo = zg[1536 + tid];
      float cn = sigm(zff) * c[tid] + sigm(zi) * tanhf(zgg);
      float hn = sigm(zo) * tanhf(cn);
      c[tid] = cn; h[tid] = hn;
      hs[(size_t)t * 512 + tid] = hn;
    }
    __syncthreads();
  }
}

// ---------------- decoder: persistent multi-block with grid barrier ---------
__device__ __forceinline__ void gridbar(int* cnt, int* gen, int nb) {
  __syncthreads();
  if (threadIdx.x == 0) {
    __threadfence();
    int g = __hip_atomic_load(gen, __ATOMIC_RELAXED, __HIP_MEMORY_SCOPE_AGENT);
    int a = __hip_atomic_fetch_add(cnt, 1, __ATOMIC_ACQ_REL, __HIP_MEMORY_SCOPE_AGENT);
    if (a == nb - 1) {
      __hip_atomic_store(cnt, 0, __ATOMIC_RELAXED, __HIP_MEMORY_SCOPE_AGENT);
      __hip_atomic_fetch_add(gen, 1, __ATOMIC_RELEASE, __HIP_MEMORY_SCOPE_AGENT);
    } else {
      while (__hip_atomic_load(gen, __ATOMIC_ACQUIRE, __HIP_MEMORY_SCOPE_AGENT) == g)
        __builtin_amdgcn_s_sleep(2);
    }
  }
  __syncthreads();
}

// scalar state indices in sst[]
// 0:last 1:wait 2:stat 3:done 4:start 5:fb 6:fbrow
__global__ __launch_bounds__(TBDEC) void k_decoder(
    const float* __restrict__ enc, const float* __restrict__ kenc,
    const float* __restrict__ attWq, const float* __restrict__ attv,
    const float* __restrict__ mpW, const float* __restrict__ mpb,
    const float* __restrict__ d1Wih, const float* __restrict__ d1Whh, const float* __restrict__ d1b,
    const float* __restrict__ d2Wih, const float* __restrict__ d2Whh, const float* __restrict__ d2b,
    const float* __restrict__ hW, const float* __restrict__ hBias,
    const float* __restrict__ omW, const float* __restrict__ omb,
    float* __restrict__ mgc_out,
    float* __restrict__ h1, float* __restrict__ c1,
    float* __restrict__ h2, float* __restrict__ c2,
    float* __restrict__ xin, float* __restrict__ zg,
    float* __restrict__ dh, float* __restrict__ aq,
    float* __restrict__ sw /*16: 0..13 weights, 15 vf*/,
    float* __restrict__ lastm, int* __restrict__ sst,
    int* __restrict__ barCnt, int* __restrict__ barGen) {
  const int tid  = threadIdx.x;
  const int gtid = blockIdx.x * TBDEC + tid;
  const int nb   = gridDim.x;

  // ---- initial state: dec1 cell with x=0,h=0,c=0 -> z = b
  if (gtid < 1024) {
    float zi = d1b[gtid], zgg = d1b[2048 + gtid], zo = d1b[3072 + gtid];
    float cn = sigm(zi) * tanhf(zgg);
    c1[gtid] = cn;
    h1[gtid] = sigm(zo) * tanhf(cn);
  }
  if (gtid < 80) lastm[gtid] = 0.f;
  if (gtid == 0) { sst[0] = 0; sst[1] = 0; sst[2] = 0; sst[3] = 0; }
  gridbar(barCnt, barGen, nb);
  // dec2 cell with x=h1, h=0, c=0
  zg[gtid] = d2b[gtid] + dot4(d2Wih + (size_t)gtid * 1024, h1, 256);
  gridbar(barCnt, barGen, nb);
  if (gtid < 1024) {
    float zi = zg[gtid], zgg = zg[2048 + gtid], zo = zg[3072 + gtid];
    float cn = sigm(zi) * tanhf(zgg);
    c2[gtid] = cn;
    h2[gtid] = sigm(zo) * tanhf(cn);
  }
  gridbar(barCnt, barGen, nb);

  for (int step = 0; step < cTDEC; ++step) {
    // ---- attention (block 0 only: aq -> scores -> softmax/flags)
    if (blockIdx.x == 0) {
      if (tid < 128) aq[tid] = dot4(attWq + (size_t)tid * 1024, c2, 256);
      if (tid == 0) {
        int st = sst[0] - 7;
        if (st < 0) st = 0;
        if (st > cS - cWIN) st = cS - cWIN;
        sst[4] = st;
      }
      __syncthreads();
      if (tid < cWIN) {
        const float* kr = kenc + (size_t)(sst[4] + tid) * 128;
        float acc = 0.f;
        for (int k = 0; k < 128; ++k) acc += tanhf(kr[k] + aq[k]) * attv[k];
        sw[tid] = acc;
      }
      __syncthreads();
      if (tid == 0) {
        float mx = sw[0]; int bi = 0;
        for (int j = 1; j < cWIN; ++j) if (sw[j] > mx) { mx = sw[j]; bi = j; }
        float ssum = 0.f;
        for (int j = 0; j < cWIN; ++j) { float e = __expf(sw[j] - mx); sw[j] = e; ssum += e; }
        float inv = 1.f / ssum;
        for (int j = 0; j < cWIN; ++j) sw[j] *= inv;
        int new_idx = bi + sst[4];
        int wait2 = (new_idx == sst[0]) ? (sst[1] + 1) : 0;
        sst[5] = (wait2 == 10) ? 1 : 0;
        int fr = sst[0] + 1; if (fr > cS - 1) fr = cS - 1;
        sst[6] = fr;
        int stat2 = sst[2] + ((new_idx >= cS - 2) ? 1 : 0);
        int valid = (!sst[3]) && (stat2 < 4);
        sw[15] = valid ? 1.f : 0.f;
        sst[0] = new_idx; sst[1] = wait2; sst[2] = stat2;
        if (!valid) sst[3] = 1;
      }
    }
    gridbar(barCnt, barGen, nb);
    // ---- context + mgc projection -> xin[0..1251]
    if (gtid < cDENC) {
      float v;
      if (sst[5]) v = enc[(size_t)sst[6] * cDENC + gtid];
      else {
        int st = sst[4];
        v = 0.f;
        for (int q = 0; q < cWIN; ++q) v += sw[q] * enc[(size_t)(st + q) * cDENC + gtid];
      }
      xin[gtid] = v;
    } else if (gtid < cDENC + cPROJ) {
      int r = gtid - cDENC;
      float acc = mpb[r] + dot4(mpW + (size_t)r * cMGC, lastm, 20);
      xin[gtid] = tanhf(acc);
    }
    gridbar(barCnt, barGen, nb);
    // ---- dec1 gates: one row per thread (4096)
    zg[gtid] = d1b[gtid] + dot4(d1Wih + (size_t)gtid * cDIN, xin, 313)
                         + dot4(d1Whh + (size_t)gtid * 1024, h1, 256);
    gridbar(barCnt, barGen, nb);
    if (gtid < 1024) {
      float zi = zg[gtid], zff = zg[1024 + gtid], zgg = zg[2048 + gtid], zo = zg[3072 + gtid];
      float cn = sigm(zff) * c1[gtid] + sigm(zi) * tanhf(zgg);
      c1[gtid] = cn;
      h1[gtid] = sigm(zo) * tanhf(cn);
    }
    gridbar(barCnt, barGen, nb);
    // ---- dec2 gates
    zg[gtid] = d2b[gtid] + dot4(d2Wih + (size_t)gtid * 1024, h1, 256)
                         + dot4(d2Whh + (size_t)gtid * 1024, h2, 256);
    gridbar(barCnt, barGen, nb);
    if (gtid < 1024) {
      float zi = zg[gtid], zff = zg[1024 + gtid], zgg = zg[2048 + gtid], zo = zg[3072 + gtid];
      float cn = sigm(zff) * c2[gtid] + sigm(zi) * tanhf(zgg);
      c2[gtid] = cn;
      h2[gtid] = sigm(zo) * tanhf(cn);
    }
    gridbar(barCnt, barGen, nb);
    // ---- dh = dec2hid_W @ h2 + b
    if (gtid < 1024) dh[gtid] = hBias[gtid] + dot4(hW + (size_t)gtid * 1024, h2, 256);
    gridbar(barCnt, barGen, nb);
    // ---- output mgc (block 0, 240 rows); last 80 feed back
    if (blockIdx.x == 0 && tid < 240) {
      float o = sigm(omb[tid] + dot4(omW + (size_t)tid * 1024, dh, 256));
      mgc_out[(size_t)step * 240 + tid] = o * sw[15];
      if (tid >= 160) lastm[tid - 160] = o;
    }
    gridbar(barCnt, barGen, nb);
  }
}

// ---------------- postnet glue ----------------
__global__ void k_x0(const float* __restrict__ mgc, unsigned short* __restrict__ x0) {
  int i = blockIdx.x * blockDim.x + threadIdx.x; // 80*1536
  if (i >= cMGC * cTPN) return;
  int ch = i / cTPN, t = i % cTPN;
  x0[i] = f2bf(mgc[(size_t)t * cMGC + ch]);
}

__global__ void k_final(const float* __restrict__ mgc, const float* __restrict__ y3,
                        float* __restrict__ out) {
  int i = blockIdx.x * blockDim.x + threadIdx.x; // 1536*80
  if (i >= cTPN * cMGC) return;
  int t = i / cMGC, ch = i % cMGC;
  out[i] = mgc[i] + y3[(size_t)ch * cTPN + t];
}

// ---------------------------------------------------------------------------
extern "C" void kernel_launch(void* const* d_in, const int* in_sizes, int n_in,
                              void* d_out, int out_size, void* d_ws, size_t ws_size,
                              hipStream_t stream) {
  (void)in_sizes; (void)n_in; (void)out_size; (void)ws_size;
  const int*   x     = (const int*)d_in[0];
  const int*   xc    = (const int*)d_in[1];
  const int*   spkid = (const int*)d_in[2];
  const float* styp  = (const float*)d_in[3];
  const float* cemb  = (const float*)d_in[4];
  const float* caemb = (const float*)d_in[5];
  const float* semb  = (const float*)d_in[6];
  const float* gst   = (const float*)d_in[7];
  const float* efWih = (const float*)d_in[8];
  const float* efWhh = (const float*)d_in[9];
  const float* efb   = (const float*)d_in[10];
  const float* ebWih = (const float*)d_in[11];
  const float* ebWhh = (const float*)d_in[12];
  const float* ebb   = (const float*)d_in[13];
  const float* mpW   = (const float*)d_in[14];
  const float* mpb   = (const float*)d_in[15];
  const float* d1Wih = (const float*)d_in[16];
  const float* d1Whh = (const float*)d_in[17];
  const float* d1b   = (const float*)d_in[18];
  const float* d2Wih = (const float*)d_in[19];
  const float* d2Whh = (const float*)d_in[20];
  const float* d2b   = (const float*)d_in[21];
  const float* hW    = (const float*)d_in[22];
  const float* hBias = (const float*)d_in[23];
  const float* omW   = (const float*)d_in[24];
  const float* omb   = (const float*)d_in[25];
  const float* attWq = (const float*)d_in[28];
  const float* attWk = (const float*)d_in[29];
  const float* attv  = (const float*)d_in[30];
  const float* pn1W  = (const float*)d_in[31];
  const float* pn1b  = (const float*)d_in[32];
  const float* pn2W  = (const float*)d_in[33];
  const float* pn2b  = (const float*)d_in[34];
  const float* pn3W  = (const float*)d_in[35];
  const float* pn3b  = (const float*)d_in[36];

  char* ws = (char*)d_ws;
  size_t off = 0;
  auto wsa = [&](size_t bytes) -> void* {
    void* p = ws + off;
    off += (bytes + 255) & ~(size_t)255;
    return p;
  };
  float*          styspk = (float*)wsa(128 * 4);
  unsigned short* xsB    = (unsigned short*)wsa((size_t)cS * 128 * 2);
  unsigned short* wihfB  = (unsigned short*)wsa((size_t)2048 * 128 * 2);
  unsigned short* wihbB  = (unsigned short*)wsa((size_t)2048 * 128 * 2);
  float*          Zf     = (float*)wsa((size_t)2048 * cS * 4);
  float*          Zb     = (float*)wsa((size_t)2048 * cS * 4);
  float*          hf     = (float*)wsa((size_t)cS * 512 * 4);
  float*          hb     = (float*)wsa((size_t)cS * 512 * 4);
  float*          enc    = (float*)wsa((size_t)cS * cDENC * 4);
  unsigned short* encB   = (unsigned short*)wsa((size_t)cS * cDENC * 2);
  unsigned short* wkB    = (unsigned short*)wsa((size_t)cWATT * cDENC * 2);
  float*          kenc   = (float*)wsa((size_t)cS * cWATT * 4);
  float*          mgcSeq = (float*)wsa((size_t)cTPN * cMGC * 4);
  unsigned short* x0B    = (unsigned short*)wsa((size_t)cMGC * cTPN * 2);
  unsigned short* pn1B   = (unsigned short*)wsa((size_t)cPNC * cKP1 * 2);
  unsigned short* pn2B   = (unsigned short*)wsa((size_t)cPNC * 1280 * 2);
  unsigned short* pn3B   = (unsigned short*)wsa((size_t)cMGC * 1280 * 2);
  unsigned short* xcol   = (unsigned short*)wsa((size_t)cTPN * 1280 * 2); // reused
  unsigned short* y1B    = (unsigned short*)wsa((size_t)cPNC * cTPN * 2);
  unsigned short* y2B    = (unsigned short*)wsa((size_t)cPNC * cTPN * 2);
  float*          y3     = (float*)wsa((size_t)cMGC * cTPN * 4);
  // decoder state
  float* h1v  = (float*)wsa(1024 * 4);
  float* c1v  = (float*)wsa(1024 * 4);
  float* h2v  = (float*)wsa(1024 * 4);
  float* c2v  = (float*)wsa(1024 * 4);
  float* xinv = (float*)wsa(1280 * 4);
  float* zgv  = (float*)wsa(4096 * 4);
  float* dhv  = (float*)wsa(1024 * 4);
  float* aqv  = (float*)wsa(128 * 4);
  float* swv  = (float*)wsa(16 * 4);
  float* lm   = (float*)wsa(80 * 4);
  int*   sst  = (int*)wsa(8 * 4);
  int*   bar  = (int*)wsa(2 * 4);

  auto grid1 = [](int n) { return dim3((unsigned)((n + 255) / 256)); };

  // ---- prep / precision conversion
  k_styspk<<<1, 128, 0, stream>>>(styp, gst, semb, spkid, styspk);
  k_build_xs<<<grid1(cS * 128), 256, 0, stream>>>(x, xc, cemb, caemb, xsB);
  k_cvt_bf16<<<grid1(2048 * 128), 256, 0, stream>>>(efWih, wihfB, 2048 * 128);
  k_cvt_bf16<<<grid1(2048 * 128), 256, 0, stream>>>(ebWih, wihbB, 2048 * 128);
  k_cvt_bf16<<<grid1(cWATT * cDENC), 256, 0, stream>>>(attWk, wkB, cWATT * cDENC);
  k_cvt_pad<<<grid1(cPNC * cKP1), 256, 0, stream>>>(pn1W, pn1B, cPNC, 400, cKP1);
  k_cvt_bf16<<<grid1(cPNC * 1280), 256, 0, stream>>>(pn2W, pn2B, cPNC * 1280);
  k_cvt_bf16<<<grid1(cMGC * 1280), 256, 0, stream>>>(pn3W, pn3B, cMGC * 1280);

  // ---- encoder input projections hoisted out of the scan (WMMA)
  {
    dim3 g(2048 / 16, cS / 16);
    wmma_gemm_bf16<<<g, 32, 0, stream>>>(wihfB, xsB, efb, Zf, nullptr, 2048, cS, 128, 0);
    wmma_gemm_bf16<<<g, 32, 0, stream>>>(wihbB, xsB, ebb, Zb, nullptr, 2048, cS, 128, 0);
  }
  // ---- bidirectional encoder recurrence (fwd+bwd concurrent)
  k_enc_scan<<<2, 1024, 0, stream>>>(efWhh, ebWhh, Zf, Zb, hf, hb);
  k_build_enc<<<grid1(cS * cDENC), 256, 0, stream>>>(hf, hb, styspk, enc, encB);

  // ---- attention keys for all positions (WMMA): kenc = enc @ att_Wk^T
  {
    dim3 g(cS / 16, cWATT / 16);
    wmma_gemm_bf16<<<g, 32, 0, stream>>>(encB, wkB, nullptr, kenc, nullptr, cS, cWATT, cDENC, 0);
  }

  // ---- autoregressive decoder: persistent 8-block grid, atomic grid barrier
  k_zero_i32<<<1, 32, 0, stream>>>(bar, 2);
  k_decoder<<<NBDEC, TBDEC, 0, stream>>>(enc, kenc, attWq, attv, mpW, mpb,
                                         d1Wih, d1Whh, d1b, d2Wih, d2Whh, d2b,
                                         hW, hBias, omW, omb, mgcSeq,
                                         h1v, c1v, h2v, c2v, xinv, zgv, dhv,
                                         aqv, swv, lm, sst, bar, bar + 1);

  // ---- postnet: three k=5 convs as im2col + WMMA GEMM (Bt layout)
  k_x0<<<grid1(cMGC * cTPN), 256, 0, stream>>>(mgcSeq, x0B);
  k_im2col<<<grid1(cTPN * cKP1), 256, 0, stream>>>(x0B, xcol, cMGC, cTPN, cKP1);
  {
    dim3 g(cPNC / 16, cTPN / 16);
    wmma_gemm_bf16<<<g, 32, 0, stream>>>(pn1B, xcol, pn1b, nullptr, y1B, cPNC, cTPN, cKP1, 1);
  }
  k_im2col<<<grid1(cTPN * 1280), 256, 0, stream>>>(y1B, xcol, cPNC, cTPN, 1280);
  {
    dim3 g(cPNC / 16, cTPN / 16);
    wmma_gemm_bf16<<<g, 32, 0, stream>>>(pn2B, xcol, pn2b, nullptr, y2B, cPNC, cTPN, 1280, 1);
  }
  k_im2col<<<grid1(cTPN * 1280), 256, 0, stream>>>(y2B, xcol, cPNC, cTPN, 1280);
  {
    dim3 g(cMGC / 16, cTPN / 16);
    wmma_gemm_bf16<<<g, 32, 0, stream>>>(pn3B, xcol, pn3b, y3, nullptr, cMGC, cTPN, 1280, 0);
  }
  k_final<<<grid1(cTPN * cMGC), 256, 0, stream>>>(mgcSeq, y3, (float*)d_out);
}
